// RNNLanguageModel_89644557402953
// MI455X (gfx1250) — compile-verified, matrix operand order checked by reading, not run on hardware
//
#include <hip/hip_runtime.h>

#define SEQ   2048
#define EMB   512
#define HID   1024
#define VOCAB 32000

#define VT_PER_CHUNK 80   // 16-wide vocab tiles per chunk -> 1280 columns
#define NCHUNK       25   // 25 * 1280 = 32000

typedef __attribute__((ext_vector_type(16))) __bf16 v16bf;
typedef __attribute__((ext_vector_type(8)))  float  v8f;

#if defined(__AMDGCN__) && __has_builtin(__builtin_amdgcn_tensor_load_to_lds)
#define USE_TDM 1
#else
#define USE_TDM 0
#endif

__device__ __forceinline__ unsigned short f32_bf16(float f) {
    unsigned u = __builtin_bit_cast(unsigned, f);
    unsigned r = u + 0x7FFFu + ((u >> 16) & 1u);   // round-to-nearest-even
    return (unsigned short)(r >> 16);
}

// Load one 16x32 bf16 WMMA operand fragment (A or B side).
// Layout (wave32): lane = row%16 ; half h=lane>>4 selects K groups:
//   vector elems 0..7  <- K = kb + h*8 .. +7        (contiguous 16B)
//   vector elems 8..15 <- K = kb + 16 + h*8 .. +7   (contiguous 16B)
__device__ __forceinline__ v16bf load_frag(const unsigned short* base,
                                           int row, int ld, int kb, int h) {
    const unsigned short* p0 = base + row * ld + kb + h * 8;
    union { v16bf v; uint4 q[2]; } u;
    u.q[0] = *(const uint4*)p0;
    u.q[1] = *(const uint4*)(p0 + 16);
    return u.v;
}

#if USE_TDM
typedef __attribute__((ext_vector_type(4))) unsigned int v4u_;
typedef __attribute__((ext_vector_type(8))) int          v8i_;
typedef __attribute__((ext_vector_type(4))) int          v4i_;

// TDM DMA: copy one 16-row x 512-col bf16 tile of E into LDS.
// D# per CDNA5 ISA ch.8: 2-D tensor (groups 2/3 zero), data_size=2B.
__device__ __forceinline__ void tdm_load_tile(const unsigned short* gsrc, unsigned lds_off) {
    unsigned long long ga = (unsigned long long)(size_t)gsrc;
    v4u_ g0;
    g0.x = 1u;                                                // count=1, user mode
    g0.y = lds_off;                                           // lds_addr (bytes)
    g0.z = (unsigned)(ga & 0xFFFFFFFFu);                      // global_addr[31:0]
    g0.w = (unsigned)((ga >> 32) & 0x1FFFFFFu) | (2u << 30);  // global_addr[56:32] | type=2
    v8i_ g1;
    g1[0] = (1 << 16);                                        // data_size=1 (2 bytes)
    g1[1] = (int)((unsigned)(EMB & 0xFFFF) << 16);            // tensor_dim0[15:0]
    g1[2] = (int)(((unsigned)EMB >> 16) | ((unsigned)(VOCAB & 0xFFFF) << 16)); // dim0 hi | dim1 lo
    g1[3] = (int)(((unsigned)VOCAB >> 16) | ((unsigned)EMB << 16));            // dim1 hi | tile_dim0=512
    g1[4] = 16;                                               // tile_dim1=16 rows, tile_dim2=0
    g1[5] = EMB;                                              // tensor_dim0_stride low32
    g1[6] = 0;
    g1[7] = 0;
    v4i_ z = {0, 0, 0, 0};
#if defined(__clang_major__) && __clang_major__ >= 23
    v8i_ z8 = {0, 0, 0, 0, 0, 0, 0, 0};
    __builtin_amdgcn_tensor_load_to_lds(g0, g1, z, z, z8, 0);
#else
    __builtin_amdgcn_tensor_load_to_lds(g0, g1, z, z, 0);
#endif
}
#endif

// ---------- prep kernels ----------

__global__ void convert_bf(const float* __restrict__ in, unsigned short* __restrict__ out, long n) {
    long i = (long)blockIdx.x * blockDim.x + threadIdx.x;
    long stride = (long)gridDim.x * blockDim.x;
    for (; i < n; i += stride) out[i] = f32_bf16(in[i]);
}

// Wt[N][K] = bf16(W[K][N])
__global__ void transpose_bf(const float* __restrict__ W, unsigned short* __restrict__ Wt,
                             int K, int N) {
    long i = (long)blockIdx.x * blockDim.x + threadIdx.x;
    long total = (long)K * N;
    if (i < total) {
        int n = (int)(i / K), k = (int)(i % K);
        Wt[(long)n * K + k] = f32_bf16(W[(long)k * N + n]);
    }
}

// Xbf[t] = bf16(E[tokens[t-1]]) ; Xbf[0] = 0   (teacher-forced shift)
__global__ void embed_shift(const int* __restrict__ tokens, const float* __restrict__ E,
                            unsigned short* __restrict__ Xbf) {
    int t = blockIdx.x;
    if (t == 0) {
        for (int k = threadIdx.x; k < EMB; k += 128) Xbf[k] = 0;
    } else {
        const float* src = E + (long)tokens[t - 1] * EMB;
        for (int k = threadIdx.x; k < EMB; k += 128)
            Xbf[t * EMB + k] = f32_bf16(src[k]);
    }
}

// ---------- XW1 = X @ W1 via WMMA (parallel over time) ----------

__global__ void xw1_wmma(const unsigned short* __restrict__ Xbf,   // [SEQ][EMB] bf16
                         const unsigned short* __restrict__ W1t,   // [HID][EMB] bf16 (transposed)
                         float* __restrict__ XW1) {                // [SEQ][HID]
    int wave = threadIdx.x >> 5;
    int lane = threadIdx.x & 31;
    int tile = blockIdx.x * 8 + wave;          // 128 row-tiles * 64 col-tiles = 8192
    int mt = tile >> 6;
    int nt = tile & 63;
    int r = lane & 15, h = lane >> 4;
    v8f acc = {};
#pragma unroll
    for (int kb = 0; kb < EMB; kb += 32) {
        v16bf a = load_frag(Xbf, mt * 16 + r, EMB, kb, h);
        v16bf b = load_frag(W1t, nt * 16 + r, EMB, kb, h);
        acc = __builtin_amdgcn_wmma_f32_16x16x32_bf16(false, a, false, b, (short)0, acc,
                                                      false, false);
    }
#pragma unroll
    for (int g = 0; g < 8; ++g)
        XW1[(mt * 16 + g + 8 * h) * HID + nt * 16 + r] = acc[g];
}

// ---------- sequential RNN scan (one workgroup, h-state in LDS) ----------

__global__ void __launch_bounds__(1024) rnn_scan(
    const float* __restrict__ XW1, const float* __restrict__ U1, const float* __restrict__ b1,
    const float* __restrict__ W2, const float* __restrict__ U2, const float* __restrict__ b2,
    float* __restrict__ H2f, unsigned short* __restrict__ H2bf) {
    __shared__ float h1s[HID];
    __shared__ float h2s[EMB];
    int j = threadIdx.x;
    h1s[j] = 0.f;
    if (j < EMB) h2s[j] = 0.f;
    __syncthreads();
    for (int t = 0; t < SEQ; ++t) {
        if (t + 1 < SEQ) __builtin_prefetch(&XW1[(t + 1) * HID + j], 0, 0);
        // layer 1: h1 = tanh(XW1[t] + h1 @ U1 + b1)   (coalesced column reads)
        float acc = XW1[t * HID + j] + b1[j];
#pragma unroll 8
        for (int k = 0; k < HID; ++k)
            acc = fmaf(h1s[k], U1[k * HID + j], acc);
        float h1n = tanhf(acc);
        __syncthreads();           // all reads of old h1 done
        h1s[j] = h1n;
        __syncthreads();
        // layer 2: h2 = tanh(h1 @ W2 + h2 @ U2 + b2)
        float h2n = 0.f;
        if (j < EMB) {
            float a2 = b2[j];
#pragma unroll 8
            for (int k = 0; k < HID; ++k)
                a2 = fmaf(h1s[k], W2[k * EMB + j], a2);
#pragma unroll 8
            for (int k = 0; k < EMB; ++k)
                a2 = fmaf(h2s[k], U2[k * EMB + j], a2);
            h2n = tanhf(a2);
        }
        __syncthreads();           // all reads of old h2 done
        if (j < EMB) {
            h2s[j] = h2n;
            H2f[t * EMB + j]  = h2n;
            H2bf[t * EMB + j] = f32_bf16(h2n);
        }
        __syncthreads();
    }
}

// ---------- fused logits GEMM + online logsumexp ----------
// Block = 8 waves, each wave owns one 16-row tile (128 seq rows per block).
// The 16x512 bf16 E tile is DMA'd into LDS by the TDM (double-buffered,
// s_wait_tensorcnt) and shared by all 8 waves.

__global__ void __launch_bounds__(256) logits_lse(
        const unsigned short* __restrict__ H2bf, // [SEQ][EMB] bf16
        const unsigned short* __restrict__ Ebf,  // [VOCAB][EMB] bf16
        float* __restrict__ pmax,                // [SEQ][NCHUNK]
        float* __restrict__ psum) {              // [SEQ][NCHUNK]
    __shared__ unsigned short lbuf[2][16 * EMB];   // double-buffered E tile (2 x 16KB)
    __shared__ float sm[8][32][8];
    __shared__ float ss[8][32][8];
    int wave = threadIdx.x >> 5, lane = threadIdx.x & 31;
    int chunk = blockIdx.y;
    int r = lane & 15, h = lane >> 4;
    int myrow0 = (blockIdx.x * 8 + wave) * 16;     // this wave's 16 sequence rows

    // hoist A fragments: 16 K-steps * v16bf, reused for every vocab tile
    v16bf afrag[16];
#pragma unroll
    for (int ks = 0; ks < 16; ++ks)
        afrag[ks] = load_frag(H2bf, myrow0 + r, EMB, ks * 32, h);

    float m[8], s[8];
#pragma unroll
    for (int g = 0; g < 8; ++g) { m[g] = -INFINITY; s[g] = 0.f; }

    int vt0 = chunk * VT_PER_CHUNK;
#if USE_TDM
    if (wave == 0)
        tdm_load_tile(Ebf + (size_t)vt0 * 16 * EMB, (unsigned)(size_t)&lbuf[0][0]);
#endif
    for (int i = 0; i < VT_PER_CHUNK; ++i) {
        int cur = i & 1;
#if USE_TDM
        if (wave == 0) {
            if (i + 1 < VT_PER_CHUNK) {
                tdm_load_tile(Ebf + (size_t)(vt0 + i + 1) * 16 * EMB,
                              (unsigned)(size_t)&lbuf[cur ^ 1][0]);
                __builtin_amdgcn_s_wait_tensorcnt(1);   // tile i landed
            } else {
                __builtin_amdgcn_s_wait_tensorcnt(0);
            }
        }
        __syncthreads();                                // publish tile i
#else
        {   // cooperative fallback copy of tile i
            const uint4* src = (const uint4*)(Ebf + (size_t)(vt0 + i) * 16 * EMB);
            uint4* dst = (uint4*)&lbuf[cur][0];
            for (int idx = threadIdx.x; idx < 16 * EMB / 8; idx += 256) dst[idx] = src[idx];
        }
        __syncthreads();
#endif
        v8f acc = {};
#pragma unroll
        for (int ks = 0; ks < 16; ++ks) {
            v16bf b = load_frag(&lbuf[cur][0], r, EMB, ks * 32, h);  // ds_load_b128
            acc = __builtin_amdgcn_wmma_f32_16x16x32_bf16(false, afrag[ks], false, b,
                                                          (short)0, acc, false, false);
        }
#pragma unroll
        for (int g = 0; g < 8; ++g) {   // online logsumexp per tracked row
            float d = acc[g];
            float mn = fmaxf(m[g], d);
            s[g] = s[g] * __expf(m[g] - mn) + __expf(d - mn);
            m[g] = mn;
        }
        __syncthreads();   // compute on lbuf[cur] done before DMA i+2 overwrites it
    }
#pragma unroll
    for (int g = 0; g < 8; ++g) { sm[wave][lane][g] = m[g]; ss[wave][lane][g] = s[g]; }
    __syncthreads();
    if (threadIdx.x < 128) {            // one thread per (wave, row-in-tile)
        int w = threadIdx.x >> 4;
        int rr = threadIdx.x & 15;
        int g = rr & 7, hh = rr >> 3;
        float M = -INFINITY, S = 0.f;
        for (int l = hh * 16; l < hh * 16 + 16; ++l) {
            float m2 = sm[w][l][g], s2 = ss[w][l][g];
            float Mn = fmaxf(M, m2);
            S = S * __expf(M - Mn) + s2 * __expf(m2 - Mn);
            M = Mn;
        }
        int row = (blockIdx.x * 8 + w) * 16 + rr;
        pmax[row * NCHUNK + chunk] = M;
        psum[row * NCHUNK + chunk] = S;
    }
}

// ---------- per-row finalize: merge chunks, target logit, log-prob ----------

__global__ void row_finalize(const float* __restrict__ pmax, const float* __restrict__ psum,
                             const float* __restrict__ H2f, const float* __restrict__ E,
                             const int* __restrict__ tokens, float* __restrict__ logprob) {
    __shared__ float red[256];
    int t = blockIdx.x;
    const float* hrow = H2f + (long)t * EMB;
    const float* erow = E + (long)tokens[t] * EMB;
    float a = 0.f;
    for (int k = threadIdx.x; k < EMB; k += 256) a = fmaf(hrow[k], erow[k], a);
    red[threadIdx.x] = a;
    __syncthreads();
    for (int off = 128; off > 0; off >>= 1) {
        if (threadIdx.x < off) red[threadIdx.x] += red[threadIdx.x + off];
        __syncthreads();
    }
    if (threadIdx.x == 0) {
        float M = -INFINITY, S = 0.f;
        for (int c = 0; c < NCHUNK; ++c) {
            float m2 = pmax[t * NCHUNK + c], s2 = psum[t * NCHUNK + c];
            float Mn = fmaxf(M, m2);
            S = S * __expf(M - Mn) + s2 * __expf(m2 - Mn);
            M = Mn;
        }
        logprob[t] = red[0] - (M + logf(S));
    }
}

__global__ void final_sum(const float* __restrict__ logprob, float* __restrict__ out) {
    __shared__ float red[256];
    float a = 0.f;
    for (int i = threadIdx.x; i < SEQ; i += 256) a += logprob[i];
    red[threadIdx.x] = a;
    __syncthreads();
    for (int off = 128; off > 0; off >>= 1) {
        if (threadIdx.x < off) red[threadIdx.x] += red[threadIdx.x + off];
        __syncthreads();
    }
    if (threadIdx.x == 0) out[0] = red[0];
}

// ---------- launcher ----------

extern "C" void kernel_launch(void* const* d_in, const int* in_sizes, int n_in,
                              void* d_out, int out_size, void* d_ws, size_t ws_size,
                              hipStream_t stream) {
    const int*   tokens = (const int*)d_in[0];
    const float* E      = (const float*)d_in[1];
    const float* W1     = (const float*)d_in[2];
    const float* U1     = (const float*)d_in[3];
    const float* b1     = (const float*)d_in[4];
    const float* W2     = (const float*)d_in[5];
    const float* U2     = (const float*)d_in[6];
    const float* b2     = (const float*)d_in[7];
    float* out = (float*)d_out;

    char* ws = (char*)d_ws;
    size_t off = 0;
    auto alloc = [&](size_t bytes) -> void* {
        void* p = ws + off;
        off += (bytes + 255) & ~(size_t)255;
        return p;
    };
    unsigned short* Ebf  = (unsigned short*)alloc((size_t)VOCAB * EMB * 2); // 32.8 MB
    unsigned short* Xbf  = (unsigned short*)alloc((size_t)SEQ * EMB * 2);
    unsigned short* W1t  = (unsigned short*)alloc((size_t)HID * EMB * 2);
    float*          XW1  = (float*)alloc((size_t)SEQ * HID * 4);
    float*          H2f  = (float*)alloc((size_t)SEQ * EMB * 4);
    unsigned short* H2bf = (unsigned short*)alloc((size_t)SEQ * EMB * 2);
    float*          pmax = (float*)alloc((size_t)SEQ * NCHUNK * 4);
    float*          psum = (float*)alloc((size_t)SEQ * NCHUNK * 4);
    float*          lprb = (float*)alloc((size_t)SEQ * 4);

    convert_bf<<<4096, 256, 0, stream>>>(E, Ebf, (long)VOCAB * EMB);
    embed_shift<<<SEQ, 128, 0, stream>>>(tokens, E, Xbf);
    transpose_bf<<<(EMB * HID + 255) / 256, 256, 0, stream>>>(W1, W1t, EMB, HID);
    xw1_wmma<<<1024, 256, 0, stream>>>(Xbf, W1t, XW1);
    rnn_scan<<<1, 1024, 0, stream>>>(XW1, U1, b1, W2, U2, b2, H2f, H2bf);
    dim3 g(SEQ / 128, NCHUNK);       // 16 x 25 blocks, 8 waves each
    logits_lse<<<g, 256, 0, stream>>>(H2bf, Ebf, pmax, psum);
    row_finalize<<<SEQ, 256, 0, stream>>>(pmax, psum, H2f, E, tokens, lprb);
    final_sum<<<1, 256, 0, stream>>>(lprb, out);
}
